// AttentionBlock_56478819942486
// MI455X (gfx1250) — matrix operand order repaired and checked by reading, not compiled, not run on hardware
//
#include <hip/hip_runtime.h>
#include <hip/hip_bf16.h>

typedef __attribute__((ext_vector_type(16))) _Float16 v16h;
typedef __attribute__((ext_vector_type(8)))  _Float16 v8h;
typedef __attribute__((ext_vector_type(8)))  float    v8f;
typedef int v4i __attribute__((vector_size(16)));   // 16B payload type for async copies

#define B_   8
#define C_   128
#define HW_  4096
#define GRP_ 32
#define EPS_ 1e-5f
#define KB_  64          // keys staged per chunk in attention

#define AS1 __attribute__((address_space(1)))
#define AS3 __attribute__((address_space(3)))

// ---------------------------------------------------------------------------
// CDNA5 async global->LDS copy (16B per lane), ASYNCcnt-tracked
// ---------------------------------------------------------------------------
static __device__ __forceinline__ void async_cp16(const _Float16* src, _Float16* dst) {
#if defined(__has_builtin) && __has_builtin(__builtin_amdgcn_global_load_async_to_lds_b128)
  __builtin_amdgcn_global_load_async_to_lds_b128((AS1 v4i*)src, (AS3 v4i*)dst, 0, 0);
#else
  asm volatile("global_load_async_to_lds_b128 %0, %1, off"
               :: "v"((unsigned)(size_t)dst), "v"(src) : "memory");
#endif
}

#if defined(__has_builtin) && __has_builtin(__builtin_amdgcn_s_wait_asynccnt)
#define WAIT_ASYNCCNT(n) __builtin_amdgcn_s_wait_asynccnt(n)
#else
#define WAIT_ASYNCCNT(n) asm volatile("s_wait_asynccnt %0" :: "n"(n) : "memory")
#endif

// ---------------------------------------------------------------------------
// WMMA helpers (CDNA5: D = A(16x32 f16) * B(32x16 f16) + C(16x16 f32))
// ---------------------------------------------------------------------------
static __device__ __forceinline__ v8f wmma_f16(v16h a, v16h b, v8f c) {
  return __builtin_amdgcn_wmma_f32_16x16x32_f16(
      /*neg_a=*/false, a, /*neg_b=*/false, b,
      /*c_mod=*/(short)0, c, /*reuse_a=*/false, /*reuse_b=*/false);
}

// A-matrix tile (16x32 f16, row-major, leading dim ld):
// lane m (<16): K 0..7 & 16..23 of row m;  lane m+16: K 8..15 & 24..31
static __device__ __forceinline__ v16h load_A16(const _Float16* __restrict__ base, int ld) {
  const int lane = threadIdx.x & 31;
  const _Float16* p = base + (lane & 15) * ld + ((lane & 16) ? 8 : 0);
  union { v16h v; v8h h[2]; } u;
  u.h[0] = *(const v8h*)(p);
  u.h[1] = *(const v8h*)(p + 16);
  return u.v;
}

// B-matrix tile (32x16 f16, row-major KxN, leading dim ld): lane l holds column N=l%16,
// K 0..7 & 16..23 (l<16) or K 8..15 & 24..31 (l>=16)
static __device__ __forceinline__ v16h load_B16(const _Float16* __restrict__ base, int ld) {
  const int lane = threadIdx.x & 31;
  const _Float16* p = base + (lane & 15) + ((lane & 16) ? 8 : 0) * ld;
  v16h v;
#pragma unroll
  for (int s = 0; s < 8; ++s) { v[s] = p[s * ld]; v[s + 8] = p[(s + 16) * ld]; }
  return v;
}

// B-tile whose per-lane row pointer is already resolved (LDS staging buffers)
static __device__ __forceinline__ v16h load_B16_row(const _Float16* p) {
  union { v16h v; v8h h[2]; } u;
  u.h[0] = *(const v8h*)(p);
  u.h[1] = *(const v8h*)(p + 8);
  return u.v;
}

// ---------------------------------------------------------------------------
// 1) GroupNorm -> normed f32 in d_out (residual base) + f16 tokens x[B][N][C]
// ---------------------------------------------------------------------------
__global__ void gn_kernel(const float* __restrict__ x, const float* __restrict__ gw,
                          const float* __restrict__ gb, float* __restrict__ normed,
                          _Float16* __restrict__ xh) {
  const int b = blockIdx.x >> 5;
  const int g = blockIdx.x & 31;
  const int cpg = C_ / GRP_;                    // 4
  const int elems = cpg * HW_;                  // 16384
  const float* base = x + ((size_t)b * C_ + (size_t)g * cpg) * HW_;

  float s = 0.f, ss = 0.f;
  for (int i = threadIdx.x; i < elems; i += blockDim.x) {
    float v = base[i]; s += v; ss += v * v;
  }
  for (int m = 16; m; m >>= 1) { s += __shfl_xor(s, m, 32); ss += __shfl_xor(ss, m, 32); }
  __shared__ float rs_[8], rss_[8];
  const int wave = threadIdx.x >> 5;
  if ((threadIdx.x & 31) == 0) { rs_[wave] = s; rss_[wave] = ss; }
  __syncthreads();
  s = 0.f; ss = 0.f;
#pragma unroll
  for (int w = 0; w < 8; ++w) { s += rs_[w]; ss += rss_[w]; }
  const float mean = s / (float)elems;
  const float var  = ss / (float)elems - mean * mean;
  const float rstd = rsqrtf(var + EPS_);

  float*    onorm = normed + ((size_t)b * C_ + (size_t)g * cpg) * HW_;
  _Float16* xt    = xh + (size_t)b * HW_ * C_;
  for (int i = threadIdx.x; i < elems; i += blockDim.x) {
    const int c = g * cpg + (i >> 12);
    const int n = i & (HW_ - 1);
    const float y = (base[i] - mean) * rstd * gw[c] + gb[c];
    onorm[i] = y;
    xt[(size_t)n * C_ + c] = (_Float16)y;
  }
}

// ---------------------------------------------------------------------------
// 2) Weight prep: W[out,in] f32 -> W^T[in,out] f16 (WMMA B-matrix source)
// ---------------------------------------------------------------------------
__global__ void wprep_kernel(const float* __restrict__ Wq, const float* __restrict__ Wk,
                             const float* __restrict__ Wv, const float* __restrict__ Wp,
                             _Float16* __restrict__ wT) {
  const int idx = blockIdx.x * blockDim.x + threadIdx.x;   // 0..65535
  const int w  = idx >> 14;
  const int e  = idx & 16383;
  const int co = e >> 7, ci = e & 127;
  const float* src = (w == 0) ? Wq : (w == 1) ? Wk : (w == 2) ? Wv : Wp;
  wT[(size_t)w * C_ * C_ + (size_t)ci * C_ + co] = (_Float16)src[e];
}

// ---------------------------------------------------------------------------
// 3) QKV projection (wave = 16-token tile); scale folded into q; k stored
//    transposed [C][N] so score B-tiles are contiguous.
// ---------------------------------------------------------------------------
__global__ void qkv_kernel(const _Float16* __restrict__ xh, const _Float16* __restrict__ wT,
                           const float* __restrict__ bq, const float* __restrict__ bk,
                           const float* __restrict__ bv,
                           _Float16* __restrict__ qh, _Float16* __restrict__ kTmat,
                           _Float16* __restrict__ vh) {
  const int lane = threadIdx.x & 31;
  const int wave = threadIdx.x >> 5;
  const int tile = blockIdx.x * 8 + wave;          // 0..2047
  const int b  = tile >> 8;
  const int n0 = (tile & 255) * 16;
  const float scale = 0.08838834764831845f;        // C^-0.5

  const _Float16* X = xh + (size_t)b * HW_ * C_ + (size_t)n0 * C_;
  v16h xa[4];
#pragma unroll
  for (int kc = 0; kc < 4; ++kc) xa[kc] = load_A16(X + kc * 32, C_);

  const int colN = lane & 15;
  const int half = lane >> 4;
  const float* biases[3] = { bq, bk, bv };

#pragma unroll
  for (int o = 0; o < 3; ++o) {
    const _Float16* W = wT + (size_t)o * C_ * C_;
#pragma unroll
    for (int t = 0; t < 8; ++t) {
      v8f acc = {};
#pragma unroll
      for (int kc = 0; kc < 4; ++kc)
        acc = wmma_f16(xa[kc], load_B16(W + (size_t)(kc * 32) * C_ + t * 16, C_), acc);
      const float bias = biases[o][t * 16 + colN];
      if (o == 1) {                       // k -> transposed layout [C][N]
        v8h pk;
#pragma unroll
        for (int r = 0; r < 8; ++r) pk[r] = (_Float16)(acc[r] + bias);
        *(v8h*)(kTmat + (size_t)b * C_ * HW_ + (size_t)(t * 16 + colN) * HW_ + n0 + 8 * half) = pk;
      } else {
        _Float16* dst = (o == 0 ? qh : vh) + (size_t)b * HW_ * C_;
        const float mul = (o == 0) ? scale : 1.0f;
#pragma unroll
        for (int r = 0; r < 8; ++r)
          dst[(size_t)(n0 + r + 8 * half) * C_ + t * 16 + colN] = (_Float16)((acc[r] + bias) * mul);
      }
    }
  }
}

// ---------------------------------------------------------------------------
// 4) Flash attention with async double-buffered K/V staging in LDS.
//    Block = 8 waves, all on the SAME batch -> K/V chunks staged once and
//    shared.  Per 64-key chunk: async copies (ASYNCcnt) overlap the previous
//    chunk's 32 WMMAs.  Online softmax on the f32 D-layout
//    (VGPR r / lane-half h = row r+8h, lane&15 = col).
// ---------------------------------------------------------------------------
__global__ void __launch_bounds__(256)
attn_kernel(const _Float16* __restrict__ qh, const _Float16* __restrict__ kTmat,
            const _Float16* __restrict__ vh, _Float16* __restrict__ ah) {
  const int tid  = threadIdx.x;
  const int lane = tid & 31;
  const int wave = tid >> 5;
  const int b    = blockIdx.x >> 5;                       // 32 blocks per batch
  const int n0   = (((blockIdx.x & 31) << 3) + wave) * 16;
  const int colN = lane & 15;
  const int half = lane >> 4;

  const _Float16* Q = qh    + (size_t)b * HW_ * C_;
  const _Float16* K = kTmat + (size_t)b * C_ * HW_;       // [C][N]
  const _Float16* V = vh    + (size_t)b * HW_ * C_;       // [N][C]

  __shared__ __align__(16) _Float16 Ks[2][C_][KB_];       // 2 x 16KB
  __shared__ __align__(16) _Float16 Vs[2][KB_][C_];       // 2 x 16KB
  __shared__ __align__(16) _Float16 Plds[8][16 * 32];     // 8KB (per-wave)
  _Float16* P = Plds[wave];

  // stage one 64-key chunk: 8 x b128 async copies per thread-slot
  auto stage = [&](int buf, int k0) {
#pragma unroll
    for (int t = 0; t < 4; ++t) {                         // K: 128 rows x 128B
      const int chunk = tid + t * 256;                    // 0..1023
      const int row = chunk >> 3, seg = chunk & 7;
      async_cp16(K + (size_t)row * HW_ + k0 + seg * 8, &Ks[buf][row][seg * 8]);
    }
#pragma unroll
    for (int t = 0; t < 4; ++t) {                         // V: 64 rows x 256B
      const int chunk = tid + t * 256;
      const int row = chunk >> 4, seg = chunk & 15;
      async_cp16(V + (size_t)(k0 + row) * C_ + seg * 8, &Vs[buf][row][seg * 8]);
    }
  };

  v16h qa[4];
#pragma unroll
  for (int kc = 0; kc < 4; ++kc) qa[kc] = load_A16(Q + (size_t)n0 * C_ + kc * 32, C_);

  float mrow[8], lrow[8];
  v8f acc[8];
#pragma unroll
  for (int r = 0; r < 8; ++r) { mrow[r] = -1e30f; lrow[r] = 0.f; }
#pragma unroll
  for (int t = 0; t < 8; ++t) acc[t] = (v8f){};

  stage(0, 0);                                            // prologue
  const int NKB = HW_ / KB_;                              // 64 chunks
  for (int kb = 0; kb < NKB; ++kb) {
    const int buf = kb & 1;
    if (kb + 1 < NKB) { stage(buf ^ 1, (kb + 1) * KB_); WAIT_ASYNCCNT(8); }
    else              { WAIT_ASYNCCNT(0); }
    __syncthreads();                                      // chunk `buf` ready

#pragma unroll
    for (int g = 0; g < 2; ++g) {                         // two 32-key groups
      v8f s0 = {}, s1 = {};
#pragma unroll
      for (int kc = 0; kc < 4; ++kc) {
        s0 = wmma_f16(qa[kc], load_B16(&Ks[buf][kc * 32][g * 32], KB_),      s0);
        s1 = wmma_f16(qa[kc], load_B16(&Ks[buf][kc * 32][g * 32 + 16], KB_), s1);
      }
      float alpha[8];
#pragma unroll
      for (int r = 0; r < 8; ++r) {
        float v = fmaxf(s0[r], s1[r]);
        v = fmaxf(v, __shfl_xor(v, 1, 32));
        v = fmaxf(v, __shfl_xor(v, 2, 32));
        v = fmaxf(v, __shfl_xor(v, 4, 32));
        v = fmaxf(v, __shfl_xor(v, 8, 32));
        const float mn = fmaxf(mrow[r], v);
        alpha[r] = __expf(mrow[r] - mn);
        mrow[r] = mn;
        const float p0 = __expf(s0[r] - mn);
        const float p1 = __expf(s1[r] - mn);
        float rs = p0 + p1;
        rs += __shfl_xor(rs, 1, 32);
        rs += __shfl_xor(rs, 2, 32);
        rs += __shfl_xor(rs, 4, 32);
        rs += __shfl_xor(rs, 8, 32);
        lrow[r] = lrow[r] * alpha[r] + rs;
        const int row = r + 8 * half;
        P[row * 32 + colN]      = (_Float16)p0;
        P[row * 32 + 16 + colN] = (_Float16)p1;
      }
#pragma unroll
      for (int t = 0; t < 8; ++t)
#pragma unroll
        for (int r = 0; r < 8; ++r) acc[t][r] *= alpha[r];

      __builtin_amdgcn_wave_barrier();                    // order P store vs load
      const v16h pa = load_A16(P, 32);                    // P as A-matrix (16x32)
#pragma unroll
      for (int t = 0; t < 8; ++t)
        acc[t] = wmma_f16(pa, load_B16(&Vs[buf][g * 32][t * 16], C_), acc[t]);
      __builtin_amdgcn_wave_barrier();
    }
    __syncthreads();                                      // protect buf^1 rewrite
  }

  _Float16* O = ah + (size_t)b * HW_ * C_;
#pragma unroll
  for (int t = 0; t < 8; ++t)
#pragma unroll
    for (int r = 0; r < 8; ++r)
      O[(size_t)(n0 + r + 8 * half) * C_ + t * 16 + colN] = (_Float16)(acc[t][r] / lrow[r]);
}

// ---------------------------------------------------------------------------
// 5) Output projection + bias accumulated onto normed residual in d_out
// ---------------------------------------------------------------------------
__global__ void proj_kernel(const _Float16* __restrict__ ah, const _Float16* __restrict__ wpT,
                            const float* __restrict__ bp, float* __restrict__ out) {
  const int lane = threadIdx.x & 31;
  const int wave = threadIdx.x >> 5;
  const int tile = blockIdx.x * 8 + wave;
  const int b  = tile >> 8;
  const int n0 = (tile & 255) * 16;
  const int colN = lane & 15;
  const int half = lane >> 4;

  const _Float16* A = ah + (size_t)b * HW_ * C_ + (size_t)n0 * C_;
  v16h aa[4];
#pragma unroll
  for (int kc = 0; kc < 4; ++kc) aa[kc] = load_A16(A + kc * 32, C_);

#pragma unroll
  for (int t = 0; t < 8; ++t) {
    v8f acc = {};
#pragma unroll
    for (int kc = 0; kc < 4; ++kc)
      acc = wmma_f16(aa[kc], load_B16(wpT + (size_t)(kc * 32) * C_ + t * 16, C_), acc);
    const float bias = bp[t * 16 + colN];
    float* o = out + (size_t)b * C_ * HW_ + (size_t)(t * 16 + colN) * HW_ + n0 + 8 * half;
#pragma unroll
    for (int r = 0; r < 8; ++r) o[r] += acc[r] + bias;
  }
}

// ---------------------------------------------------------------------------
extern "C" void kernel_launch(void* const* d_in, const int* in_sizes, int n_in,
                              void* d_out, int out_size, void* d_ws, size_t ws_size,
                              hipStream_t stream) {
  const float* x  = (const float*)d_in[0];
  const float* gw = (const float*)d_in[1];
  const float* gb = (const float*)d_in[2];
  const float* Wq = (const float*)d_in[3];
  const float* bq = (const float*)d_in[4];
  const float* Wk = (const float*)d_in[5];
  const float* bk = (const float*)d_in[6];
  const float* Wv = (const float*)d_in[7];
  const float* bv = (const float*)d_in[8];
  const float* Wp = (const float*)d_in[9];
  const float* bp = (const float*)d_in[10];
  float* out = (float*)d_out;

  const size_t tokElems = (size_t)B_ * HW_ * C_;      // 4,194,304 per tensor
  _Float16* xh  = (_Float16*)d_ws;                    // tokens; reused as attn output
  _Float16* qh  = xh  + tokElems;
  _Float16* kT  = qh  + tokElems;
  _Float16* vh  = kT  + tokElems;
  _Float16* wT  = vh  + tokElems;                     // Wq^T,Wk^T,Wv^T,Wp^T f16
  _Float16* wpT = wT  + (size_t)3 * C_ * C_;
  // total ws use: 4*8MB + 128KB ~= 33.7 MB

  wprep_kernel<<<256, 256, 0, stream>>>(Wq, Wk, Wv, Wp, wT);
  gn_kernel  <<<B_ * GRP_, 256, 0, stream>>>(x, gw, gb, out, xh);
  qkv_kernel <<<256, 256, 0, stream>>>(xh, wT, bq, bk, bv, qh, kT, vh);
  attn_kernel<<<256, 256, 0, stream>>>(qh, kT, vh, /*ah=*/xh);
  proj_kernel<<<256, 256, 0, stream>>>(/*ah=*/xh, wpT, bp, out);
}
